// SelfAttentionUnit_32315333935894
// MI455X (gfx1250) — matrix operand order repaired and checked by reading, not compiled, not run on hardware
//
#include <hip/hip_runtime.h>

// ---------------------------------------------------------------------------
// Fused self-attention unit for MI455X (gfx1250, wave32, WMMA + TDM).
// Heavy GEMMs run on v_wmma_f32_16x16x32_bf16 (bf16 in, f32 accumulate).
// The kv = k*v^T GEMM stages tiles into LDS with tensor_load_to_lds (TDM).
// ---------------------------------------------------------------------------

typedef __attribute__((ext_vector_type(16))) __bf16 bf16x16;
typedef __attribute__((ext_vector_type(8)))  __bf16 bf16x8;
typedef __attribute__((ext_vector_type(8)))  float  f32x8;
typedef __attribute__((ext_vector_type(4)))  unsigned int u32x4;
typedef __attribute__((ext_vector_type(8)))  unsigned int u32x8;

namespace {
constexpr int B_   = 16;
constexpr int CIN  = 131;
constexpr int KP   = 160;   // Cin padded to a multiple of 32 for K of WMMA
constexpr int N_   = 4096;
constexpr int C_   = 128;
constexpr int H_   = 256;   // 2C
constexpr int HQKV = 768;   // 3*H stacked q,k,v rows
constexpr int SPL  = 8;     // split-K factor for the kv GEMM (K = 4096)
constexpr int LDS_PITCH = 136;  // 128 bf16 + TDM pad (4 dwords) per row
constexpr float EPSV = 1e-5f;

constexpr size_t al(size_t x) { return (x + 255) & ~(size_t)255; }
constexpr size_t SZ_WQKV = (size_t)HQKV * KP * 2;
constexpr size_t SZ_WF   = (size_t)C_ * H_ * 2;
constexpr size_t SZ_SQ   = (size_t)HQKV * 4;
constexpr size_t SZ_SF   = (size_t)C_ * 4;
constexpr size_t SZ_XT   = (size_t)B_ * N_ * KP * 2;
constexpr size_t SZ_QT   = (size_t)B_ * N_ * H_ * 2;
constexpr size_t SZ_KV   = (size_t)B_ * H_ * N_ * 2;
constexpr size_t SZ_KVP  = (size_t)B_ * SPL * H_ * H_ * 4;
constexpr size_t SZ_KVT  = (size_t)B_ * H_ * H_ * 2;

constexpr size_t OFF_WQKV = 0;
constexpr size_t OFF_WF   = al(OFF_WQKV + SZ_WQKV);
constexpr size_t OFF_SQ   = al(OFF_WF   + SZ_WF);
constexpr size_t OFF_BQ   = al(OFF_SQ   + SZ_SQ);
constexpr size_t OFF_SF   = al(OFF_BQ   + SZ_SQ);
constexpr size_t OFF_BF   = al(OFF_SF   + SZ_SF);
constexpr size_t OFF_XT   = al(OFF_BF   + SZ_SF);
constexpr size_t OFF_QT   = al(OFF_XT   + SZ_XT);
constexpr size_t OFF_K    = al(OFF_QT   + SZ_QT);
constexpr size_t OFF_V    = al(OFF_K    + SZ_KV);
constexpr size_t OFF_KVP  = al(OFF_V    + SZ_KV);
constexpr size_t OFF_KVT  = al(OFF_KVP  + SZ_KVP);
constexpr size_t OFF_VALT = al(OFF_KVT  + SZ_KVT);   // total ~191 MB of d_ws
} // namespace

// ---- WMMA fragment helpers (contiguous k = 16*(lane/16) + s for both A and B, wave32) ----------

// A fragment: 16(M) x 32(K) bf16 from a row-major matrix (lda in elements).
// lane 0-15: row M=lane, K = k0+0..15
// lane 16-31: row M=lane-16, K = k0+16..31
__device__ __forceinline__ bf16x16 frag_a_rm(const __bf16* A, size_t lda,
                                             int m0, int k0, int lane) {
  const int row = m0 + (lane & 15);
  const int kb  = k0 + ((lane >> 4) << 4);
  const __bf16* p = A + (size_t)row * lda + kb;
  bf16x8 lo = *(const bf16x8*)(p);        // 16B load
  bf16x8 hi = *(const bf16x8*)(p + 8);    // 16B load
  bf16x16 r;
#pragma unroll
  for (int i = 0; i < 8; ++i) { r[i] = lo[i]; r[i + 8] = hi[i]; }
  return r;
}

// B fragment: 32(K) x 16(N) bf16 from an N-major buffer BT[n][k] (ldb elems):
// lane 0-15 = column n0+lane, K=k0..k0+15; lane 16-31, K=k0+16..k0+31.
__device__ __forceinline__ bf16x16 frag_b_nm(const __bf16* BT, size_t ldb,
                                             int k0, int n0, int lane) {
  const int col = n0 + (lane & 15);
  const int kb  = k0 + ((lane >> 4) << 4);
  return *(const bf16x16*)(BT + (size_t)col * ldb + kb);  // 32B contiguous
}

// Same as frag_b_nm but split into two 16B loads (LDS pitch is 16B-, not
// 32B-aligned). Used for LDS-resident tiles -> ds_load_b128.
__device__ __forceinline__ bf16x16 frag_b_nm16(const __bf16* BT, size_t ldb,
                                               int k0, int n0, int lane) {
  const int col = n0 + (lane & 15);
  const int kb  = k0 + ((lane >> 4) << 4);
  const __bf16* p = BT + (size_t)col * ldb + kb;
  bf16x8 lo = *(const bf16x8*)(p);
  bf16x8 hi = *(const bf16x8*)(p + 8);
  bf16x16 r;
#pragma unroll
  for (int i = 0; i < 8; ++i) { r[i] = lo[i]; r[i + 8] = hi[i]; }
  return r;
}

__device__ __forceinline__ f32x8 wmma_bf16(bf16x16 a, bf16x16 b, f32x8 c) {
  return __builtin_amdgcn_wmma_f32_16x16x32_bf16(
      /*neg_a=*/false, a, /*neg_b=*/false, b,
      /*c_mod=*/(short)0, c, /*reuse_a=*/false, /*reuse_b=*/false);
}

// ---- Tensor Data Mover: 2-D tile load, global(bf16) -> LDS -----------------
// Builds the D# (groups 0/1 per CDNA5 ISA 8.3/8.4) in SGPRs and issues the
// 2-group form of tensor_load_to_lds (VADDR2/3 = NULL: tensor up to 2D).
// Pad config inserts 4 dwords after every 64 dwords -> LDS row pitch 136 bf16.
__device__ __forceinline__ void tdm_load_2d(unsigned int lds_off,
                                            unsigned long long gaddr,
                                            int tile_k, int tile_rows,
                                            int tensor_d0, int tensor_d1,
                                            int stride0) {
  u32x4 g0;
  g0[0] = 1u;                                             // count=1, user mode
  g0[1] = lds_off;                                        // LDS byte address
  g0[2] = (unsigned int)(gaddr & 0xFFFFFFFFull);          // global_addr[31:0]
  g0[3] = (unsigned int)((gaddr >> 32) & 0x1FFFFFFull)    // global_addr[56:32]
        | (2u << 30);                                     // type = 2 (image)
  u32x8 g1;
  g1[0] = (1u << 16)        // data_size = 1 (2 bytes)
        | (1u << 20)        // pad_enable
        | (5u << 22)        // pad_interval: 64 dwords
        | (3u << 25);       // pad_amount:   4 dwords
  g1[1] = ((unsigned int)tensor_d0 & 0xFFFFu) << 16;      // tensor_dim0[15:0]
  g1[2] = (((unsigned int)tensor_d0 >> 16) & 0xFFFFu)     // tensor_dim0[31:16]
        | (((unsigned int)tensor_d1 & 0xFFFFu) << 16);    // tensor_dim1[15:0]
  g1[3] = (((unsigned int)tensor_d1 >> 16) & 0xFFFFu)     // tensor_dim1[31:16]
        | ((unsigned int)tile_k << 16);                   // tile_dim0
  g1[4] = (unsigned int)tile_rows;                        // tile_dim1 (dim2=0)
  g1[5] = (unsigned int)stride0;                          // dim0_stride[31:0]
  g1[6] = 0u;                                             // stride bits / dim1_stride
  g1[7] = 0u;
  asm volatile("tensor_load_to_lds %0, %1" :: "s"(g0), "s"(g1) : "memory");
}

// ---- prep kernels ----------------------------------------------------------

__global__ void k_prep_w(const float* __restrict__ wq, const float* __restrict__ wk,
                         const float* __restrict__ wv, __bf16* __restrict__ WQKV) {
  const int i = blockIdx.x * blockDim.x + threadIdx.x;
  if (i >= HQKV * KP) return;
  const int row = i / KP, col = i % KP;
  float v = 0.f;
  if (col < CIN) {
    if (row < H_)          v = wq[row * CIN + col];
    else if (row < 2 * H_) v = wk[(row - H_) * CIN + col];
    else                   v = wv[(row - 2 * H_) * CIN + col];
  }
  WQKV[i] = (__bf16)v;
}

__global__ void k_prep_wf(const float* __restrict__ wf, __bf16* __restrict__ WF) {
  const int i = blockIdx.x * blockDim.x + threadIdx.x;
  if (i >= C_ * H_) return;
  WF[i] = (__bf16)wf[i];
}

__global__ void k_prep_sb(const float* gq, const float* bq, const float* mq, const float* vq,
                          const float* gk, const float* bk, const float* mk, const float* vk,
                          const float* gv, const float* bv, const float* mv, const float* vv,
                          const float* gf, const float* bff, const float* mf, const float* vf,
                          float* SQ, float* BQ, float* SF, float* BF) {
  const int i = blockIdx.x * blockDim.x + threadIdx.x;
  if (i < HQKV) {
    const int sel = i >> 8, j = i & 255;
    const float* g = (sel == 0) ? gq : (sel == 1) ? gk : gv;
    const float* b = (sel == 0) ? bq : (sel == 1) ? bk : bv;
    const float* m = (sel == 0) ? mq : (sel == 1) ? mk : mv;
    const float* v = (sel == 0) ? vq : (sel == 1) ? vk : vv;
    const float s = g[j] / sqrtf(v[j] + EPSV);
    SQ[i] = s;
    BQ[i] = b[j] - m[j] * s;
  } else if (i < HQKV + C_) {
    const int j = i - HQKV;
    const float s = gf[j] / sqrtf(vf[j] + EPSV);
    SF[j] = s;
    BF[j] = bff[j] - mf[j] * s;
  }
}

// LDS-tiled transpose+convert: x[b][c][n] f32 -> xT[b][n][c] bf16, c padded
__global__ __launch_bounds__(256) void k_xpose(const float* __restrict__ x,
                                               __bf16* __restrict__ XT) {
  __shared__ float tile[32][33];
  const int c0 = blockIdx.x * 32;       // 5 tiles over KP=160
  const int n0 = blockIdx.y * 32;       // 128 tiles over N
  const int b  = blockIdx.z;
  const int tx = threadIdx.x, ty = threadIdx.y;
#pragma unroll
  for (int cc = ty; cc < 32; cc += 8) {
    const int c = c0 + cc;
    float v = 0.f;
    if (c < CIN) v = x[((size_t)b * CIN + c) * N_ + n0 + tx];
    tile[cc][tx] = v;
  }
  __syncthreads();
#pragma unroll
  for (int nn = ty; nn < 32; nn += 8)
    XT[((size_t)b * N_ + n0 + nn) * KP + c0 + tx] = (__bf16)tile[tx][nn];
}

// ---- GEMM 1: fused QKV projection + BN + ReLU ------------------------------
__global__ __launch_bounds__(256) void k_qkv(const __bf16* __restrict__ W,
    const __bf16* __restrict__ XT, const float* __restrict__ SQ,
    const float* __restrict__ BQ, __bf16* __restrict__ QT,
    __bf16* __restrict__ Kb, __bf16* __restrict__ Vb) {
  const int lane = threadIdx.x & 31;
  const int wid  = (blockIdx.x * blockDim.x + threadIdx.x) >> 5;
  const int perb = (HQKV / 16) * (N_ / 64);
  const int b  = wid / perb;
  const int t  = wid % perb;
  const int m0 = (t / (N_ / 64)) * 16;
  const int n0 = (t % (N_ / 64)) * 64;
  const __bf16* xt = XT + (size_t)b * N_ * KP;

  f32x8 acc[4] = {};
  for (int k0 = 0; k0 < KP; k0 += 32) {
    const bf16x16 a = frag_a_rm(W, KP, m0, k0, lane);
#pragma unroll
    for (int j = 0; j < 4; ++j)
      acc[j] = wmma_bf16(a, frag_b_nm(xt, KP, k0, n0 + 16 * j, lane), acc[j]);
  }

  const int hi = lane >> 4, nl = lane & 15;
  float sc[8], bi[8];
#pragma unroll
  for (int r = 0; r < 8; ++r) {
    const int m = m0 + r + 8 * hi;
    sc[r] = SQ[m]; bi[r] = BQ[m];
  }
#pragma unroll
  for (int j = 0; j < 4; ++j)
#pragma unroll
    for (int r = 0; r < 8; ++r) {
      const int m = m0 + r + 8 * hi;
      const int n = n0 + 16 * j + nl;
      float v = acc[j][r] * sc[r] + bi[r];
      v = v > 0.f ? v : 0.f;
      const __bf16 h = (__bf16)v;
      if (m0 < H_)           QT[((size_t)b * N_ + n) * H_ + m] = h;            // q transposed
      else if (m0 < 2 * H_)  Kb[((size_t)b * H_ + (m - H_)) * N_ + n] = h;     // k natural
      else                   Vb[((size_t)b * H_ + (m - 2 * H_)) * N_ + n] = h; // v natural
    }
}

// ---- GEMM 2: kv[c][d] = sum_n k[c][n]*v[d][n], split-K, TDM-staged ---------
// Block = 8 waves computes a 64(M) x 128(D) tile over a K-chunk of 512.
// Wave 0 TDM-loads a 64x128 k-tile and 128x128 v-tile into padded LDS;
// all waves consume them as WMMA fragments via ds_load_b128.
__global__ __launch_bounds__(256) void k_kv(const __bf16* __restrict__ Kb,
    const __bf16* __restrict__ Vb, float* __restrict__ KVP) {
  __shared__ __bf16 ldsK[64  * LDS_PITCH];   // 17.0 KB
  __shared__ __bf16 ldsV[128 * LDS_PITCH];   // 34.0 KB
  const int lane = threadIdx.x & 31;
  const int wv32 = threadIdx.x >> 5;         // 0..7
  const int perb = SPL * 4 * 2;              // 8 splits * 4 Mblk * 2 Dblk
  const int b    = blockIdx.x / perb;
  const int t    = blockIdx.x % perb;
  const int s    = t / 8;
  const int m0b  = ((t % 8) / 2) * 64;       // M block base (c rows of k)
  const int d0b  = (t % 2) * 128;            // D block base (rows of v)
  const int kc0  = s * (N_ / SPL);           // split-K base
  const int mSub = (wv32 >> 1) * 16;         // wave's 16-row slice
  const int dSub = (wv32 & 1) * 64;          // wave's 64-col slice

  const unsigned long long kg = (unsigned long long)(Kb + (size_t)b * H_ * N_);
  const unsigned long long vg = (unsigned long long)(Vb + (size_t)b * H_ * N_);
  const unsigned int ldsKoff = (unsigned int)(size_t)(void*)ldsK;
  const unsigned int ldsVoff = (unsigned int)(size_t)(void*)ldsV;

  f32x8 acc[4] = {};
  for (int kc = 0; kc < N_ / SPL; kc += 128) {
    if (wv32 == 0) {
      tdm_load_2d(ldsKoff, kg + 2ull * ((size_t)m0b * N_ + kc0 + kc),
                  /*tile_k=*/128, /*rows=*/64, N_, H_, N_);
      tdm_load_2d(ldsVoff, vg + 2ull * ((size_t)d0b * N_ + kc0 + kc),
                  /*tile_k=*/128, /*rows=*/128, N_, H_, N_);
      __builtin_amdgcn_s_wait_tensorcnt(0);
    }
    __syncthreads();                         // tiles visible to all waves
#pragma unroll
    for (int k0 = 0; k0 < 128; k0 += 32) {
      const bf16x16 a = frag_a_rm(ldsK, LDS_PITCH, mSub, k0, lane);
#pragma unroll
      for (int j = 0; j < 4; ++j)
        acc[j] = wmma_bf16(a, frag_b_nm16(ldsV, LDS_PITCH, k0, dSub + 16 * j, lane),
                           acc[j]);
    }
    __syncthreads();                         // done reading before next TDM
  }
  const int hi = lane >> 4, nl = lane & 15;
  float* outp = KVP + ((size_t)b * SPL + s) * H_ * H_;
#pragma unroll
  for (int j = 0; j < 4; ++j)
#pragma unroll
    for (int r = 0; r < 8; ++r)
      outp[(size_t)(m0b + mSub + r + 8 * hi) * H_ + d0b + dSub + 16 * j + nl] =
          acc[j][r];
}

// deterministic split-K reduce + transpose: kvT[d][c] = sum_s KVP[b][s][c][d]
__global__ void k_kvred(const float* __restrict__ KVP, __bf16* __restrict__ KVT) {
  const size_t idx = (size_t)blockIdx.x * blockDim.x + threadIdx.x;
  const int b = (int)(idx >> 16);
  const int c = (int)((idx >> 8) & 255);
  const int d = (int)(idx & 255);
  float s = 0.f;
#pragma unroll
  for (int sp = 0; sp < SPL; ++sp)
    s += KVP[(((size_t)b * SPL + sp) * H_ + c) * H_ + d];
  KVT[((size_t)b * H_ + d) * H_ + c] = (__bf16)s;
}

// ---- GEMM 3: value[d][n] = sum_c kvT[d][c] * qT[n][c]; stored transposed ---
__global__ __launch_bounds__(256) void k_val(const __bf16* __restrict__ KVT,
    const __bf16* __restrict__ QT, __bf16* __restrict__ VALT) {
  const int lane = threadIdx.x & 31;
  const int wid  = (blockIdx.x * blockDim.x + threadIdx.x) >> 5;
  const int perb = (H_ / 16) * (N_ / 64);
  const int b  = wid / perb;
  const int t  = wid % perb;
  const int m0 = (t / (N_ / 64)) * 16;      // d
  const int n0 = (t % (N_ / 64)) * 64;      // n
  const __bf16* A  = KVT + (size_t)b * H_ * H_;
  const __bf16* BT = QT  + (size_t)b * N_ * H_;

  f32x8 acc[4] = {};
  for (int k0 = 0; k0 < H_; k0 += 32) {
    const bf16x16 a = frag_a_rm(A, H_, m0, k0, lane);
#pragma unroll
    for (int j = 0; j < 4; ++j)
      acc[j] = wmma_bf16(a, frag_b_nm(BT, H_, k0, n0 + 16 * j, lane), acc[j]);
  }
  const int hi = lane >> 4, nl = lane & 15;
#pragma unroll
  for (int j = 0; j < 4; ++j)
#pragma unroll
    for (int r = 0; r < 8; ++r)
      VALT[((size_t)b * N_ + n0 + 16 * j + nl) * H_ + m0 + r + 8 * hi] =
          (__bf16)acc[j][r];
}

// ---- GEMM 4: out[o][n] = ReLU(BN(wf[o][:] . value[:][n])) ------------------
__global__ __launch_bounds__(256) void k_out(const __bf16* __restrict__ WF,
    const __bf16* __restrict__ VALT, const float* __restrict__ SF,
    const float* __restrict__ BF, float* __restrict__ out) {
  const int lane = threadIdx.x & 31;
  const int wid  = (blockIdx.x * blockDim.x + threadIdx.x) >> 5;
  const int perb = (C_ / 16) * (N_ / 64);
  const int b  = wid / perb;
  const int t  = wid % perb;
  const int m0 = (t / (N_ / 64)) * 16;
  const int n0 = (t % (N_ / 64)) * 64;
  const __bf16* BT = VALT + (size_t)b * N_ * H_;

  f32x8 acc[4] = {};
  for (int k0 = 0; k0 < H_; k0 += 32) {
    const bf16x16 a = frag_a_rm(WF, H_, m0, k0, lane);
#pragma unroll
    for (int j = 0; j < 4; ++j)
      acc[j] = wmma_bf16(a, frag_b_nm(BT, H_, k0, n0 + 16 * j, lane), acc[j]);
  }
  const int hi = lane >> 4, nl = lane & 15;
  float sc[8], bi[8];
#pragma unroll
  for (int r = 0; r < 8; ++r) {
    const int m = m0 + r + 8 * hi;
    sc[r] = SF[m]; bi[r] = BF[m];
  }
#pragma unroll
  for (int j = 0; j < 4; ++j)
#pragma unroll
    for (int r = 0; r < 8; ++r) {
      const int m = m0 + r + 8 * hi;
      const int n = n0 + 16 * j + nl;
      float v = acc[j][r] * sc[r] + bi[r];
      out[((size_t)b * C_ + m) * N_ + n] = v > 0.f ? v : 0.f;
    }
}

// ---------------------------------------------------------------------------

extern "C" void kernel_launch(void* const* d_in, const int* in_sizes, int n_in,
                              void* d_out, int out_size, void* d_ws, size_t ws_size,
                              hipStream_t stream) {
  (void)in_sizes; (void)n_in; (void)out_size; (void)ws_size;
  const float* x  = (const float*)d_in[0];
  const float* wq = (const float*)d_in[1];
  const float* gq = (const float*)d_in[2];
  const float* bq = (const float*)d_in[3];
  const float* mq = (const float*)d_in[4];
  const float* vq = (const float*)d_in[5];
  const float* wk = (const float*)d_in[6];
  const float* gk = (const float*)d_in[7];
  const float* bk = (const float*)d_in[8];
  const float* mk = (const float*)d_in[9];
  const float* vk = (const float*)d_in[10];
  const float* wv = (const float*)d_in[11];
  const float* gv = (const float*)d_in[12];
  const float* bv = (const float*)d_in[13];
  const float* mv = (const float*)d_in[14];
  const float* vv = (const float*)d_in[15];
  const float* wf = (const float*)d_in[16];
  const float* gf = (const float*)d_in[17];
  const float* bf = (const float*)d_in[18];
  const float* mf = (const float*)d_in[19];
  const float* vf = (const float*)d_in[20];
  float* out = (float*)d_out;

  char* ws = (char*)d_ws;
  __bf16* WQKV = (__bf16*)(ws + OFF_WQKV);
  __bf16* WF   = (__bf16*)(ws + OFF_WF);
  float*  SQ   = (float*)(ws + OFF_SQ);
  float*  BQ   = (float*)(ws + OFF_BQ);
  float*  SF   = (float*)(ws + OFF_SF);
  float*  BF   = (float*)(ws + OFF_BF);
  __bf16* XT   = (__bf16*)(ws + OFF_XT);
  __bf16* QT   = (__bf16*)(ws + OFF_QT);
  __bf16* Kb   = (__bf16*)(ws + OFF_K);
  __bf16* Vb   = (__bf16*)(ws + OFF_V);
  float*  KVP  = (float*)(ws + OFF_KVP);
  __bf16* KVT  = (__bf16*)(ws + OFF_KVT);
  __bf16* VALT = (__bf16*)(ws + OFF_VALT);

  k_prep_w <<<(HQKV * KP + 255) / 256, 256, 0, stream>>>(wq, wk, wv, WQKV);
  k_prep_wf<<<(C_ * H_ + 255) / 256, 256, 0, stream>>>(wf, WF);
  k_prep_sb<<<4, 256, 0, stream>>>(gq, bq, mq, vq, gk, bk, mk, vk,
                                   gv, bv, mv, vv, gf, bf, mf, vf,
                                   SQ, BQ, SF, BF);
  {
    dim3 tb(32, 8);
    dim3 tg(KP / 32, N_ / 32, B_);
    k_xpose<<<tg, tb, 0, stream>>>(x, XT);
  }
  k_qkv  <<<B_ * (HQKV / 16) * (N_ / 64) / 8, 256, 0, stream>>>(WQKV, XT, SQ, BQ, QT, Kb, Vb);
  k_kv   <<<B_ * SPL * 4 * 2, 256, 0, stream>>>(Kb, Vb, KVP);
  k_kvred<<<B_ * H_ * H_ / 256, 256, 0, stream>>>(KVP, KVT);
  k_val  <<<B_ * (H_ / 16) * (N_ / 64) / 8, 256, 0, stream>>>(KVT, QT, VALT);
  k_out  <<<B_ * (C_ / 16) * (N_ / 64) / 8, 256, 0, stream>>>(WF, VALT, SF, BF, out);
}
